// VA_Aggregator_90829968376426
// MI455X (gfx1250) — matrix operand order repaired and checked by reading, not compiled, not run on hardware
//
#include <hip/hip_runtime.h>
#include <hip/hip_fp16.h>

typedef __attribute__((ext_vector_type(16))) _Float16 v16h;
typedef __attribute__((ext_vector_type(8)))  _Float16 v8h;
typedef __attribute__((ext_vector_type(2)))  _Float16 h2v;
typedef __attribute__((ext_vector_type(8)))  float    v8f;
typedef __attribute__((ext_vector_type(2)))  float    f2v;

#define L_HIST   50
#define D_EMB    64
#define K1       192          // 3*D concat width
#define WAVES    4
#define THREADS  (WAVES * 32)

// ---------------- LDS layout (all offsets 16B aligned) ----------------
// Shared across block:
constexpr int OFF_W1T  = 0;                          // [64][192] f16, N-major (B-fragment friendly)
constexpr int OFF_W2T  = OFF_W1T + 64 * 192 * 2;     // [64][64]  f16, N-major
constexpr int OFF_B1   = OFF_W2T + 64 * 64 * 2;      // [64] f32
constexpr int OFF_B2   = OFF_B1  + 64 * 4;           // [64] f32
constexpr int OFF_W3   = OFF_B2  + 64 * 4;           // [64] f32
constexpr int OFF_WAVE = OFF_W3  + 64 * 4;           // per-wave regions start (33536)
// Per wave:
constexpr int WB_X     = 0;                          // [64][192] f16 row buffer (x = [e_va|rep|e_af])
constexpr int WB_H1    = WB_X  + 64 * 192 * 2;       // [16][64] f16 layer-1 output tile
constexpr int WB_SC    = WB_H1 + 16 * 64 * 2;        // [64] f32 scores
constexpr int WB_ATT   = WB_SC + 64 * 4;             // [64] f32 attention weights
constexpr int WAVE_BYTES = WB_ATT + 64 * 4;          // 27136
constexpr int SMEM_BYTES = OFF_WAVE + WAVES * WAVE_BYTES;   // 142080 (< 320KB/WGP, 2 WGs fit)

// ----- WMMA fragment loaders from LDS -----
// 16-bit A 16x32 layout (contiguous, matches B): lanes 0-15 hold M=lane, halves0..15=K(kb..kb+15);
// lanes 16-31 hold M=lane-16 with K offset +16.
__device__ __forceinline__ v16h load_a_frag(const _Float16* rowbase, int lane, int kbase, int stride) {
    const int m  = lane & 15;
    const int ko = kbase + ((lane >> 4) << 4);
    const _Float16* p = rowbase + m * stride + ko;
    v8h lo = *(const v8h*)(p);         // K = ko .. ko+7
    v8h hi = *(const v8h*)(p + 8);     // K = ko+8 .. ko+15
    return __builtin_shufflevector(lo, hi, 0,1,2,3,4,5,6,7,8,9,10,11,12,13,14,15);
}
// 16-bit B 32x16: lane n = lane&15 holds column N, K = kbase + (lane<16 ? 0..15 : 16..31),
// packed linearly two-per-VGPR. wt is N-major with row stride = total K.
__device__ __forceinline__ v16h load_b_frag(const _Float16* wt, int lane, int kbase, int ntile, int stride) {
    const int n  = ntile * 16 + (lane & 15);
    const int ko = kbase + ((lane >> 4) << 4);
    const _Float16* p = wt + n * stride + ko;
    v8h lo = *(const v8h*)(p);         // K = ko .. ko+7
    v8h hi = *(const v8h*)(p + 8);     // K = ko+8 .. ko+15
    return __builtin_shufflevector(lo, hi, 0,1,2,3,4,5,6,7,8,9,10,11,12,13,14,15);
}

__global__ void __launch_bounds__(THREADS)
va_aggregator_kernel(const int* __restrict__ nodes,
                     const int* __restrict__ hva,
                     const int* __restrict__ haf,
                     const float* __restrict__ v2e,
                     const float* __restrict__ a2e,
                     const float* __restrict__ f2e,
                     const float* __restrict__ W1,
                     const float* __restrict__ b1,
                     const float* __restrict__ W2,
                     const float* __restrict__ b2,
                     const float* __restrict__ w3,
                     float* __restrict__ out,
                     int Ntot) {
    extern __shared__ char smem[];
    _Float16* w1t = (_Float16*)(smem + OFF_W1T);
    _Float16* w2t = (_Float16*)(smem + OFF_W2T);
    float*    b1s = (float*)(smem + OFF_B1);
    float*    b2s = (float*)(smem + OFF_B2);
    float*    w3s = (float*)(smem + OFF_W3);

    const int tid  = threadIdx.x;
    const int lane = tid & 31;
    const int wave = tid >> 5;

    // ---- Phase 0: stage weights (transposed, f16) + biases into LDS ----
    for (int i = tid; i < 192 * 64; i += THREADS) {
        int k = i >> 6, n = i & 63;                 // W1 is [192][64] row-major
        w1t[n * 192 + k] = (_Float16)W1[i];
    }
    for (int i = tid; i < 64 * 64; i += THREADS) {
        int k = i >> 6, n = i & 63;                 // W2 is [64][64] row-major
        w2t[n * 64 + k] = (_Float16)W2[i];
    }
    if (tid < 64) { b1s[tid] = b1[tid]; b2s[tid] = b2[tid]; w3s[tid] = w3[tid]; }
    __syncthreads();

    const int node = blockIdx.x * WAVES + wave;
    if (node >= Ntot) return;                       // no barriers past this point

    char* wbase  = smem + OFF_WAVE + wave * WAVE_BYTES;
    _Float16* xw   = (_Float16*)(wbase + WB_X);     // [64][192]
    _Float16* h1w  = (_Float16*)(wbase + WB_H1);    // [16][64]
    float*    scw  = (float*)(wbase + WB_SC);       // [64]
    float*    attw = (float*)(wbase + WB_ATT);      // [64]

    // ---- Phase 1: gather [e_va | rep | e_af] rows -> f16 LDS row buffer ----
    {
        const int it = nodes[node];
        f2v r = *(const f2v*)(v2e + (size_t)it * D_EMB + 2 * lane);
        h2v rp; rp.x = (_Float16)r.x; rp.y = (_Float16)r.y;
        for (int l = 0; l < L_HIST; ++l)
            *(h2v*)(xw + l * K1 + D_EMB + 2 * lane) = rp;
    }
    const int hbase = node * L_HIST;
    for (int l = 0; l < L_HIST; ++l) {
        const int ia = hva[hbase + l];
        const int jf = haf[hbase + l];
        if (l + 1 < L_HIST) {   // hide L2 gather latency (global_prefetch_b8)
            __builtin_prefetch(a2e + (size_t)hva[hbase + l + 1] * D_EMB + 2 * lane, 0, 0);
            __builtin_prefetch(f2e + (size_t)haf[hbase + l + 1] * D_EMB + 2 * lane, 0, 0);
        }
        f2v va = *(const f2v*)(a2e + (size_t)ia * D_EMB + 2 * lane);
        f2v af = *(const f2v*)(f2e + (size_t)jf * D_EMB + 2 * lane);
        h2v hva2; hva2.x = (_Float16)va.x; hva2.y = (_Float16)va.y;
        h2v haf2; haf2.x = (_Float16)af.x; haf2.y = (_Float16)af.y;
        *(h2v*)(xw + l * K1 + 2 * lane)           = hva2;   // e_va -> cols 0..63
        *(h2v*)(xw + l * K1 + 2 * D_EMB + 2 * lane) = haf2; // e_af -> cols 128..191
    }
    // rows 50..63 left as-is: their scores are masked out of the softmax.

    // ---- Phase 2: MLP via WMMA, 4 M-tiles of 16 rows ----
    const int nl = lane & 15;        // N (column) index inside a 16-wide tile
    const int mh = (lane >> 4) << 3; // +8 row offset for upper half-wave in C/D layout
    for (int mt = 0; mt < 4; ++mt) {
        const _Float16* xtile = xw + mt * 16 * K1;

        // Layer 1: [16x192] x [192x64]  (6 K-steps x 4 N-tiles)
        v8f acc[4] = {v8f{0}, v8f{0}, v8f{0}, v8f{0}};
        for (int kb = 0; kb < 6; ++kb) {
            v16h a = load_a_frag(xtile, lane, kb * 32, K1);
            #pragma unroll
            for (int nt = 0; nt < 4; ++nt) {
                v16h b = load_b_frag(w1t, lane, kb * 32, nt, K1);
                acc[nt] = __builtin_amdgcn_wmma_f32_16x16x32_f16(
                    false, a, false, b, (short)0, acc[nt], false, false);
            }
        }
        // bias + ReLU -> f16 h1 tile in LDS (D layout: VGPR r holds row mh+r, col nt*16+nl)
        #pragma unroll
        for (int nt = 0; nt < 4; ++nt) {
            const float bb = b1s[nt * 16 + nl];
            #pragma unroll
            for (int r = 0; r < 8; ++r) {
                float v = fmaxf(acc[nt][r] + bb, 0.0f);
                h1w[(mh + r) * 64 + nt * 16 + nl] = (_Float16)v;
            }
        }

        // Layer 2: [16x64] x [64x64]  (2 K-steps x 4 N-tiles); same-wave LDS ops are in order
        v8f acc2[4] = {v8f{0}, v8f{0}, v8f{0}, v8f{0}};
        for (int kb = 0; kb < 2; ++kb) {
            v16h a = load_a_frag(h1w, lane, kb * 32, 64);
            #pragma unroll
            for (int nt = 0; nt < 4; ++nt) {
                v16h b = load_b_frag(w2t, lane, kb * 32, nt, 64);
                acc2[nt] = __builtin_amdgcn_wmma_f32_16x16x32_f16(
                    false, a, false, b, (short)0, acc2[nt], false, false);
            }
        }

        // Scorer: relu(h2 + b2) . w3, reduced across the 16 lanes holding one row
        #pragma unroll
        for (int r = 0; r < 8; ++r) {
            float s = 0.0f;
            #pragma unroll
            for (int nt = 0; nt < 4; ++nt) {
                float v = fmaxf(acc2[nt][r] + b2s[nt * 16 + nl], 0.0f);
                s = fmaf(v, w3s[nt * 16 + nl], s);
            }
            s += __shfl_xor(s, 1);
            s += __shfl_xor(s, 2);
            s += __shfl_xor(s, 4);
            s += __shfl_xor(s, 8);
            if (nl == 0) scw[mt * 16 + mh + r] = s;   // lanes 0 and 16 write
        }
    }

    // ---- Phase 3: softmax over the 50 valid scores (b3 dropped: shift-invariant) ----
    const int r0 = 2 * lane, r1 = 2 * lane + 1;
    float s0 = (r0 < L_HIST) ? scw[r0] : -3.0e38f;
    float s1 = (r1 < L_HIST) ? scw[r1] : -3.0e38f;
    float m = fmaxf(s0, s1);
    #pragma unroll
    for (int off = 16; off >= 1; off >>= 1) m = fmaxf(m, __shfl_xor(m, off));
    float e0 = (r0 < L_HIST) ? __expf(s0 - m) : 0.0f;
    float e1 = (r1 < L_HIST) ? __expf(s1 - m) : 0.0f;
    float sum = e0 + e1;
    #pragma unroll
    for (int off = 16; off >= 1; off >>= 1) sum += __shfl_xor(sum, off);
    const float inv = 1.0f / sum;
    attw[r0] = e0 * inv;
    attw[r1] = e1 * inv;

    // ---- Phase 4: out[n] = sum_l att[l] * e_va[l]  (e_va from f16 row buffer) ----
    const int d0 = 2 * lane;
    float o0 = 0.0f, o1 = 0.0f;
    for (int l = 0; l < L_HIST; ++l) {
        const float a = attw[l];                       // LDS broadcast
        h2v e = *(const h2v*)(xw + l * K1 + d0);
        o0 = fmaf(a, (float)e.x, o0);
        o1 = fmaf(a, (float)e.y, o1);
    }
    f2v o; o.x = o0; o.y = o1;
    *(f2v*)(out + (size_t)node * D_EMB + d0) = o;
}

extern "C" void kernel_launch(void* const* d_in, const int* in_sizes, int n_in,
                              void* d_out, int out_size, void* d_ws, size_t ws_size,
                              hipStream_t stream) {
    const int*   nodes = (const int*)d_in[0];
    const int*   hva   = (const int*)d_in[1];
    const int*   haf   = (const int*)d_in[2];
    const float* v2e   = (const float*)d_in[3];
    const float* a2e   = (const float*)d_in[4];
    const float* f2e   = (const float*)d_in[5];
    const float* W1    = (const float*)d_in[6];
    const float* b1    = (const float*)d_in[7];
    const float* W2    = (const float*)d_in[8];
    const float* b2    = (const float*)d_in[9];
    const float* w3    = (const float*)d_in[10];
    // d_in[11] = b3: added uniformly to every score -> cancels in softmax; unused.
    float* out = (float*)d_out;

    const int N = in_sizes[0];
    const int blocks = (N + WAVES - 1) / WAVES;

    hipFuncSetAttribute((const void*)va_aggregator_kernel,
                        hipFuncAttributeMaxDynamicSharedMemorySize, SMEM_BYTES);
    hipLaunchKernelGGL(va_aggregator_kernel, dim3(blocks), dim3(THREADS), SMEM_BYTES, stream,
                       nodes, hva, haf, v2e, a2e, f2e, W1, b1, W2, b2, w3, out, N);
}
